// TransformerBlock_30167850287876
// MI455X (gfx1250) — matrix operand order repaired and checked by reading, not compiled, not run on hardware
//
#include <hip/hip_runtime.h>
#include <hip/hip_bf16.h>

// ---------------------------------------------------------------------------
// Types for CDNA5 WMMA
// ---------------------------------------------------------------------------
typedef __attribute__((ext_vector_type(16))) _Float16 v16h;
typedef __attribute__((ext_vector_type(8)))  _Float16 v8h;
typedef __attribute__((ext_vector_type(8)))  float    v8f;

#define LAMBDA_INIT 0.35550906759096927f

// Problem constants
#define BB      2
#define CC      256
#define SS      16
#define NN_     4096          // 16^3
#define HH      4
#define HD      64
#define PP      64
#define MROWS   8192          // B*N
#define KCONV   6912          // 27*256 (tap-major)

// ---------------------------------------------------------------------------
// Fragment loaders (contiguous k order, same for A and B)
//  A 16x32: lane m=l&15; halves 0-15 = K off+{0..15}, off = (lane>=16)*16
//  B 32x16: lane n=l&15; halves 0-15 = K off+{0..15}, off = (lane>=16)*16
// ---------------------------------------------------------------------------
__device__ __forceinline__ v16h load_a_frag(const _Float16* __restrict__ rowp,
                                            int k0, int hi)
{
    const _Float16* pa = rowp + k0 + (hi << 4);
    union { v16h v; v8h h[2]; } ua;
    ua.h[0] = *(const v8h*)(pa);
    ua.h[1] = *(const v8h*)(pa + 8);
    return ua.v;
}

// ---------------------------------------------------------------------------
// Generic GEMM via WMMA:  Out[b][M][Nc] = A[b][M][K] * B + bias
//   BT=true : B is "NT" weights, row-major [Nc][K]  (contiguous in K)
//   BT=false: B is "NN" matrix,  row-major [K][Nc]  (strided gather)
// One 16(M) x 64(N) output tile per wave (4 accumulators), K-step 32,
// software-pipelined loads.  Requires M%16==0, Nc%64==0, K%32==0.
// ---------------------------------------------------------------------------
template<bool BT>
__global__ __launch_bounds__(128, 1)
void gemm_wmma(const _Float16* __restrict__ A,
               const _Float16* __restrict__ B,
               float* __restrict__ Out,
               const float* __restrict__ bias,
               int M, int Nc, int K,
               long long sAb, long long sBb, long long sOb)
{
    const int lane = threadIdx.x & 31;
    const int wv   = threadIdx.x >> 5;
    const int nwv  = blockDim.x >> 5;
    const int tiles_m = M >> 4;
    const int tiles_n = Nc >> 6;
    const int tile = blockIdx.x * nwv + wv;
    if (tile >= tiles_m * tiles_n) return;     // wave-uniform exit

    const long long b = blockIdx.y;
    A   += b * sAb;
    B   += b * sBb;
    Out += b * sOb;

    const int tm  = tile % tiles_m;
    const int tn  = tile / tiles_m;
    const int hi  = lane >> 4;
    const int row = (tm << 4) + (lane & 15);
    const _Float16* arow = A + (long long)row * K;
    int col[4];
#pragma unroll
    for (int u = 0; u < 4; ++u) col[u] = (tn << 6) + (u << 4) + (lane & 15);

    auto load_b = [&](int k0, int u) -> v16h {
        if (BT) {
            return *(const v16h*)(B + (long long)col[u] * K + k0 + (hi << 4));
        } else {
            union { v16h v; _Float16 e[16]; } ub;
            const _Float16* pb = B + (long long)(k0 + (hi << 4)) * Nc + col[u];
#pragma unroll
            for (int i = 0; i < 16; ++i) ub.e[i] = pb[(long long)i * Nc];
            return ub.v;
        }
    };

    v8f acc[4] = {};
    v16h a_cur = load_a_frag(arow, 0, hi);
    v16h b_cur[4];
#pragma unroll
    for (int u = 0; u < 4; ++u) b_cur[u] = load_b(0, u);

    for (int k0 = 0; k0 < K; k0 += 32) {
        const int kn = (k0 + 32 < K) ? (k0 + 32) : k0;   // clamped prefetch (no branch)
        if (k0 + 64 < K) __builtin_prefetch(arow + k0 + 64, 0, 0);  // global_prefetch_b8
        v16h a_nxt = load_a_frag(arow, kn, hi);
        v16h b_nxt[4];
#pragma unroll
        for (int u = 0; u < 4; ++u) b_nxt[u] = load_b(kn, u);
#pragma unroll
        for (int u = 0; u < 4; ++u)
            acc[u] = __builtin_amdgcn_wmma_f32_16x16x32_f16(false, a_cur, false, b_cur[u],
                                                            (short)0, acc[u], false, false);
        a_cur = a_nxt;
#pragma unroll
        for (int u = 0; u < 4; ++u) b_cur[u] = b_nxt[u];
    }

#pragma unroll
    for (int u = 0; u < 4; ++u) {
        const float bv = bias ? bias[col[u]] : 0.0f;
#pragma unroll
        for (int r = 0; r < 8; ++r) {
            const int m = (tm << 4) + r + (hi << 3);
            Out[(long long)m * Nc + col[u]] = acc[u][r] + bv;
        }
    }
}

// ---------------------------------------------------------------------------
// Implicit-GEMM 3x3x3 conv (pad 1), channels-last activations.
//  In  f16 [B, 4096, 256]   (channels-last: (b*4096+n)*256 + ic)
//  W   f16 [256][27][256]   (tap-major K: k = tap*256 + ic)
//  Out f32 [B*4096, 256]
// 16(M) x 64(N) tile per wave. Flattened (tap, k-step) loop = 216 steps,
// software-pipelined; branchless zero-padding (clamped address + AND mask).
// ---------------------------------------------------------------------------
__device__ __forceinline__ void conv_load_step(const _Float16* __restrict__ In,
                                               const _Float16* __restrict__ W,
                                               int s, int b, int z, int y, int xx,
                                               const int* col, int hi,
                                               v16h& av, v16h bw[4])
{
    const int tap = s >> 3;
    const int k0  = (s & 7) << 5;
    const int dz = tap / 9;
    const int rr = tap - dz * 9;
    const int dy = rr / 3;
    const int dx = rr - dy * 3;
    const int zz = z + dz - 1, yy = y + dy - 1, xc = xx + dx - 1;
    const bool valid = ((unsigned)zz < 16u) & ((unsigned)yy < 16u) & ((unsigned)xc < 16u);
    const int nb = valid ? ((zz << 8) + (yy << 4) + xc) : 0;    // clamp to a safe row
    const _Float16* arow = In + (((long long)(b << 12) + nb) << 8);
    union { v16h v; unsigned w[8]; } ua;
    ua.v = load_a_frag(arow, k0, hi);
    const unsigned msk = valid ? 0xFFFFFFFFu : 0u;              // branchless zero-fill
#pragma unroll
    for (int i = 0; i < 8; ++i) ua.w[i] &= msk;
    av = ua.v;
    const long long wb = ((long long)tap << 8) + k0 + (hi << 4);
#pragma unroll
    for (int u = 0; u < 4; ++u)
        bw[u] = *(const v16h*)(W + (long long)col[u] * KCONV + wb);
}

__global__ __launch_bounds__(128, 1)
void conv3d_wmma(const _Float16* __restrict__ In,
                 const _Float16* __restrict__ W,
                 const float* __restrict__ bias,
                 float* __restrict__ Out)
{
    const int lane = threadIdx.x & 31;
    const int wv   = threadIdx.x >> 5;
    const int nwv  = blockDim.x >> 5;
    const int tiles_m = MROWS >> 4;                 // 512
    const int tiles_n = CC >> 6;                    // 4
    const int tile = blockIdx.x * nwv + wv;
    if (tile >= tiles_m * tiles_n) return;

    const int tm  = tile % tiles_m;
    const int tn  = tile / tiles_m;
    const int hi  = lane >> 4;
    const int row = (tm << 4) + (lane & 15);
    const int b   = row >> 12;
    const int n   = row & (NN_ - 1);
    const int z   = n >> 8, y = (n >> 4) & 15, xx = n & 15;
    int col[4];
#pragma unroll
    for (int u = 0; u < 4; ++u) col[u] = (tn << 6) + (u << 4) + (lane & 15);

    v8f acc[4] = {};
    v16h a_cur, b_cur[4], a_nxt, b_nxt[4];
    conv_load_step(In, W, 0, b, z, y, xx, col, hi, a_cur, b_cur);

    for (int s = 0; s < 216; ++s) {
        const int sn = (s + 1 < 216) ? (s + 1) : s;             // clamped tail
        conv_load_step(In, W, sn, b, z, y, xx, col, hi, a_nxt, b_nxt);
#pragma unroll
        for (int u = 0; u < 4; ++u)
            acc[u] = __builtin_amdgcn_wmma_f32_16x16x32_f16(false, a_cur, false, b_cur[u],
                                                            (short)0, acc[u], false, false);
        a_cur = a_nxt;
#pragma unroll
        for (int u = 0; u < 4; ++u) b_cur[u] = b_nxt[u];
    }

#pragma unroll
    for (int u = 0; u < 4; ++u) {
        const float bv = bias[col[u]];
#pragma unroll
        for (int r = 0; r < 8; ++r) {
            const int m = (tm << 4) + r + (hi << 3);
            Out[(long long)m * CC + col[u]] = acc[u][r] + bv;
        }
    }
}

// ---------------------------------------------------------------------------
// Small helper kernels
// ---------------------------------------------------------------------------
__global__ void cvt_f32_f16(const float* __restrict__ in, _Float16* __restrict__ out,
                            long long n)
{
    long long i = (long long)blockIdx.x * blockDim.x + threadIdx.x;
    if (i < n) out[i] = (_Float16)in[i];
}

// conv weight reorder: [oc][ic][27] f32 -> [oc][tap][ic] f16
__global__ void cvt_convw(const float* __restrict__ in, _Float16* __restrict__ out)
{
    long long i = (long long)blockIdx.x * blockDim.x + threadIdx.x;  // 256*6912
    if (i >= (long long)CC * KCONV) return;
    const int oc = (int)(i / KCONV);
    const int r  = (int)(i - (long long)oc * KCONV);
    const int t  = r >> 8;
    const int ic = r & 255;
    out[i] = (_Float16)in[(long long)oc * KCONV + ic * 27 + t];
}

__global__ void lambda_kernel(const float* lq1, const float* lk1,
                              const float* lq2, const float* lk2, float* lam)
{
    __shared__ float s1[256], s2[256];
    const int t = threadIdx.x;
    s1[t] = lq1[t] * lk1[t];
    s2[t] = lq2[t] * lk2[t];
    __syncthreads();
    for (int o = 128; o; o >>= 1) {
        if (t < o) { s1[t] += s1[t + o]; s2[t] += s2[t + o]; }
        __syncthreads();
    }
    if (t == 0) *lam = expf(s1[0]) - expf(s2[0]) + LAMBDA_INIT;
}

// x [B,C,N] f32 -> LayerNorm over C -> xln [B,N,C] f16
__global__ void ln_kernel(const float* __restrict__ x, const float* __restrict__ g,
                          const float* __restrict__ bta, _Float16* __restrict__ xln)
{
    const int bn = blockIdx.x;                  // B*N
    const int b  = bn >> 12, n = bn & (NN_ - 1);
    const int c  = threadIdx.x;                 // 256
    const float v = x[(((long long)b * CC + c) << 12) + n];
    __shared__ float sm[256], sv[256];
    sm[c] = v; sv[c] = v * v;
    __syncthreads();
    for (int o = 128; o; o >>= 1) {
        if (c < o) { sm[c] += sm[c + o]; sv[c] += sv[c + o]; }
        __syncthreads();
    }
    const float mean = sm[0] * (1.0f / 256.0f);
    const float var  = sv[0] * (1.0f / 256.0f) - mean * mean;
    const float r    = rsqrtf(var + 1e-5f);
    xln[((long long)bn << 8) + c] = (_Float16)((v - mean) * r * g[c] + bta[c]);
}

// qk_out [B*N,1024] -> qn16/kn16/k16 [B*2H,hd,N] f16 and qnT16 [B*2H,N,hd] f16
__global__ void normqk_kernel(const float* __restrict__ qk,
                              _Float16* __restrict__ qn, _Float16* __restrict__ qnT,
                              _Float16* __restrict__ kn, _Float16* __restrict__ k16)
{
    const int bd = blockIdx.x;                  // beta*64 + d  (1024)
    const int beta = bd >> 6, d = bd & 63;
    const int b = beta >> 3, h2 = beta & 7;
    const int t = threadIdx.x;
    const long long qb = ((long long)b << 12) * 1024 + h2 * 64 + d;
    const long long kb = qb + 512;
    float sq = 0.f, sk = 0.f;
    for (int n = t; n < NN_; n += 256) {
        const float q = qk[qb + ((long long)n << 10)];
        const float k = qk[kb + ((long long)n << 10)];
        sq += q * q; sk += k * k;
    }
    __shared__ float aq[256], ak[256];
    aq[t] = sq; ak[t] = sk;
    __syncthreads();
    for (int o = 128; o; o >>= 1) {
        if (t < o) { aq[t] += aq[t + o]; ak[t] += ak[t + o]; }
        __syncthreads();
    }
    const float rq = 1.0f / fmaxf(sqrtf(aq[0]), 1e-12f);
    const float rk = 1.0f / fmaxf(sqrtf(ak[0]), 1e-12f);
    for (int n = t; n < NN_; n += 256) {
        const float q = qk[qb + ((long long)n << 10)];
        const float k = qk[kb + ((long long)n << 10)];
        qn [((long long)bd << 12) + n] = (_Float16)(q * rq);
        kn [((long long)bd << 12) + n] = (_Float16)(k * rk);
        k16[((long long)bd << 12) + n] = (_Float16)k;
        qnT[((((long long)beta << 12) + n) << 6) + d] = (_Float16)(q * rq);
    }
}

// vv_out [B*N,512] -> v_ca16 / v_sa16 [B,H,hd,N] f16
__global__ void splitv_kernel(const float* __restrict__ vv,
                              _Float16* __restrict__ vca, _Float16* __restrict__ vsa)
{
    long long i = (long long)blockIdx.x * blockDim.x + threadIdx.x;  // B*H*hd*N = 2M
    if (i >= (long long)BB * HH * HD * NN_) return;
    const int n = (int)(i & (NN_ - 1));
    long long r = i >> 12;
    const int d = (int)(r & 63); r >>= 6;
    const int h = (int)(r & 3);
    const int b = (int)(r >> 2);
    const long long src = ((((long long)b << 12) + n) << 9) + h * 64 + d;
    vca[i] = (_Float16)vv[src];
    vsa[i] = (_Float16)vv[src + 256];
}

// differential softmax (channel): a [B,2H,64,64]*temp1 -> softmax(a0 - lam*a1) f16
__global__ void softmax_ca(const float* __restrict__ a, const float* __restrict__ t1,
                           const float* __restrict__ lamp, _Float16* __restrict__ out)
{
    const int rowid = blockIdx.x;               // B*H*64 = 512
    const int i = rowid & 63, h = (rowid >> 6) & 3, b = rowid >> 8;
    const int j = threadIdx.x;                  // 64
    const float lam = *lamp;
    const long long a0 = ((((long long)b * 8 + 2 * h    ) * 64 + i) << 6);
    const long long a1 = ((((long long)b * 8 + 2 * h + 1) * 64 + i) << 6);
    const float v = a[a0 + j] * t1[2 * h] - lam * a[a1 + j] * t1[2 * h + 1];
    __shared__ float sh[64];
    sh[j] = v; __syncthreads();
    for (int o = 32; o; o >>= 1) { if (j < o) sh[j] = fmaxf(sh[j], sh[j + o]); __syncthreads(); }
    const float mx = sh[0]; __syncthreads();
    const float e = expf(v - mx);
    sh[j] = e; __syncthreads();
    for (int o = 32; o; o >>= 1) { if (j < o) sh[j] += sh[j + o]; __syncthreads(); }
    out[((long long)rowid << 6) + j] = (_Float16)(e / sh[0]);
}

// differential softmax (spatial): a [B,2H,N,P]*temp2 -> softmax over P, f16 [B,H,N,P]
__global__ void softmax_sa(const float* __restrict__ a, const float* __restrict__ t2,
                           const float* __restrict__ lamp, _Float16* __restrict__ out)
{
    const int rowid = blockIdx.x;               // B*H*N = 32768
    const int n = rowid & (NN_ - 1), h = (rowid >> 12) & 3, b = rowid >> 14;
    const int p = threadIdx.x;                  // 64
    const float lam = *lamp;
    const long long a0 = (((((long long)b * 8 + 2 * h    ) << 12) + n) << 6);
    const long long a1 = (((((long long)b * 8 + 2 * h + 1) << 12) + n) << 6);
    const float v = a[a0 + p] * t2[2 * h] - lam * a[a1 + p] * t2[2 * h + 1];
    __shared__ float sh[64];
    sh[p] = v; __syncthreads();
    for (int o = 32; o; o >>= 1) { if (p < o) sh[p] = fmaxf(sh[p], sh[p + o]); __syncthreads(); }
    const float mx = sh[0]; __syncthreads();
    const float e = expf(v - mx);
    sh[p] = e; __syncthreads();
    for (int o = 32; o; o >>= 1) { if (p < o) sh[p] += sh[p + o]; __syncthreads(); }
    out[((long long)rowid << 6) + p] = (_Float16)(e / sh[0]);
}

// x_ca_raw [B,H,hd,N] -> f16 [B,N,C];  x_sa_raw [B,H,N,hd] -> f16 [B,N,C]
__global__ void xpose_ca(const float* __restrict__ xc, _Float16* __restrict__ o)
{
    long long i = (long long)blockIdx.x * blockDim.x + threadIdx.x;  // 2M over [B,N,C]
    if (i >= (long long)BB * NN_ * CC) return;
    const int cd = (int)(i & 255);
    const long long bn = i >> 8;
    const int n = (int)(bn & (NN_ - 1)), b = (int)(bn >> 12);
    const int h = cd >> 6, d = cd & 63;
    o[i] = (_Float16)xc[(((((long long)b * HH + h) << 6) + d) << 12) + n];
}
__global__ void xpose_sa(const float* __restrict__ xs, _Float16* __restrict__ o)
{
    long long i = (long long)blockIdx.x * blockDim.x + threadIdx.x;
    if (i >= (long long)BB * NN_ * CC) return;
    const int cd = (int)(i & 255);
    const long long bn = i >> 8;
    const int n = (int)(bn & (NN_ - 1)), b = (int)(bn >> 12);
    const int h = cd >> 6, d = cd & 63;
    o[i] = (_Float16)xs[((((((long long)b * HH + h) << 12) + n) << 6) + d)];
}

// attn residual: skip32 [B,C,N] f32 and skip16 channels-last [B,N,C] f16
__global__ void combine_kernel(const float* __restrict__ x, const float* __restrict__ osa,
                               const float* __restrict__ oca, const float* __restrict__ gamma,
                               float* __restrict__ skip32, _Float16* __restrict__ skip16)
{
    long long i = (long long)blockIdx.x * blockDim.x + threadIdx.x;  // [B,C,N]
    if (i >= (long long)BB * CC * NN_) return;
    const int n = (int)(i & (NN_ - 1));
    const long long t = i >> 12;
    const int c = (int)(t & 255), b = (int)(t >> 8);
    const long long bn = (((long long)b << 12) + n) << 7;            // *128
    const float e = (c < 128) ? osa[bn + c] : oca[bn + (c - 128)];
    const float v = x[i] + gamma[c] * e;
    skip32[i] = v;
    skip16[(((((long long)b << 12) + n) << 8) + c)] = (_Float16)v;
}

// per-channel BatchNorm stats over [8192,256]
__global__ void bnstats_kernel(const float* __restrict__ y, float* __restrict__ mean,
                               float* __restrict__ rstd)
{
    const int c = blockIdx.x, t = threadIdx.x;
    float s = 0.f, s2 = 0.f;
    for (int r = t; r < MROWS; r += 256) {
        const float v = y[((long long)r << 8) + c];
        s += v; s2 += v * v;
    }
    __shared__ float a[256], b[256];
    a[t] = s; b[t] = s2;
    __syncthreads();
    for (int o = 128; o; o >>= 1) { if (t < o) { a[t] += a[t + o]; b[t] += b[t + o]; } __syncthreads(); }
    if (t == 0) {
        const float m   = a[0] * (1.0f / MROWS);
        const float var = b[0] * (1.0f / MROWS) - m * m;
        mean[c] = m; rstd[c] = rsqrtf(var + 1e-5f);
    }
}

// BN + leaky_relu: [B,N,C] f32 -> [B,N,C] f16 (channels-last conv2 input)
__global__ void bnact1_kernel(const float* __restrict__ y, const float* __restrict__ mean,
                              const float* __restrict__ rstd, const float* __restrict__ g,
                              const float* __restrict__ bta, _Float16* __restrict__ act)
{
    long long i = (long long)blockIdx.x * blockDim.x + threadIdx.x;  // [B,N,C]
    if (i >= (long long)BB * NN_ * CC) return;
    const int c = (int)(i & 255);
    float v = y[i];
    v = (v - mean[c]) * rstd[c] * g[c] + bta[c];
    v = v > 0.f ? v : 0.01f * v;
    act[i] = (_Float16)v;
}

// BN2 + residual + leaky_relu -> o16 [B,N,C] f16 (input of 1x1x1 conv GEMM)
__global__ void bnact2_kernel(const float* __restrict__ y, const float* __restrict__ mean,
                              const float* __restrict__ rstd, const float* __restrict__ g,
                              const float* __restrict__ bta, const float* __restrict__ skip32,
                              _Float16* __restrict__ o16)
{
    long long i = (long long)blockIdx.x * blockDim.x + threadIdx.x;  // [B,N,C]
    if (i >= (long long)BB * NN_ * CC) return;
    const int c = (int)(i & 255);
    const long long bn = i >> 8;
    const int n = (int)(bn & (NN_ - 1)), b = (int)(bn >> 12);
    float v = y[i];
    v = (v - mean[c]) * rstd[c] * g[c] + bta[c]
        + skip32[((((long long)b << 8) + c) << 12) + n];
    v = v > 0.f ? v : 0.01f * v;
    o16[i] = (_Float16)v;
}

// out[b,c,n] = skip32[b,c,n] + conv8[b,n,c]
__global__ void final_kernel(const float* __restrict__ skip32, const float* __restrict__ c8,
                             float* __restrict__ out)
{
    long long i = (long long)blockIdx.x * blockDim.x + threadIdx.x;  // [B,C,N]
    if (i >= (long long)BB * CC * NN_) return;
    const int n = (int)(i & (NN_ - 1));
    const long long t = i >> 12;
    const int c = (int)(t & 255), b = (int)(t >> 8);
    out[i] = skip32[i] + c8[(((((long long)b << 12) + n) << 8) + c)];
}

// ---------------------------------------------------------------------------
// Host orchestration
// ---------------------------------------------------------------------------
extern "C" void kernel_launch(void* const* d_in, const int* in_sizes, int n_in,
                              void* d_out, int out_size, void* d_ws, size_t ws_size,
                              hipStream_t stream)
{
    (void)in_sizes; (void)n_in; (void)out_size; (void)ws_size;
    const float* x      = (const float*)d_in[0];
    const float* qk_w   = (const float*)d_in[1];
    const float* vv_w   = (const float*)d_in[2];
    const float* E_w    = (const float*)d_in[3];
    const float* E_b    = (const float*)d_in[4];
    const float* temp1  = (const float*)d_in[5];
    const float* temp2  = (const float*)d_in[6];
    const float* lq1    = (const float*)d_in[7];
    const float* lk1    = (const float*)d_in[8];
    const float* lq2    = (const float*)d_in[9];
    const float* lk2    = (const float*)d_in[10];
    const float* op1_w  = (const float*)d_in[11];
    const float* op1_b  = (const float*)d_in[12];
    const float* op2_w  = (const float*)d_in[13];
    const float* op2_b  = (const float*)d_in[14];
    const float* ln_g   = (const float*)d_in[15];
    const float* ln_b   = (const float*)d_in[16];
    const float* gamma  = (const float*)d_in[17];
    const float* c1_w   = (const float*)d_in[18];
    const float* c1_b   = (const float*)d_in[19];
    const float* bn1_g  = (const float*)d_in[20];
    const float* bn1_b  = (const float*)d_in[21];
    const float* c2_w   = (const float*)d_in[22];
    const float* c2_b   = (const float*)d_in[23];
    const float* bn2_g  = (const float*)d_in[24];
    const float* bn2_b  = (const float*)d_in[25];
    const float* c8_w   = (const float*)d_in[26];
    const float* c8_b   = (const float*)d_in[27];
    float* out = (float*)d_out;

    // ---- workspace bump allocator (256B aligned) ----
    char* base = (char*)d_ws;
    size_t off = 0;
    auto alloc = [&](size_t bytes) -> void* {
        off = (off + 255) & ~(size_t)255;
        void* p = base + off;
        off += bytes;
        return p;
    };
    const size_t F16 = 2, F32 = 4;
    _Float16* w_qk16  = (_Float16*)alloc(1024 * 256 * F16);
    _Float16* w_vv16  = (_Float16*)alloc(512 * 256 * F16);
    _Float16* w_E16   = (_Float16*)alloc(64 * 4096 * F16);
    _Float16* w_op1   = (_Float16*)alloc(128 * 256 * F16);
    _Float16* w_op2   = (_Float16*)alloc(128 * 256 * F16);
    _Float16* w_c1    = (_Float16*)alloc((size_t)256 * KCONV * F16);
    _Float16* w_c2    = (_Float16*)alloc((size_t)256 * KCONV * F16);
    _Float16* w_c8    = (_Float16*)alloc(256 * 256 * F16);
    float*    lam     = (float*)alloc(256);
    float*    bn1_m   = (float*)alloc(256 * F32);
    float*    bn1_r   = (float*)alloc(256 * F32);
    float*    bn2_m   = (float*)alloc(256 * F32);
    float*    bn2_r   = (float*)alloc(256 * F32);
    _Float16* xln16   = (_Float16*)alloc((size_t)MROWS * 256 * F16);       // reused: act1_16
    float*    qk_out  = (float*)alloc((size_t)MROWS * 1024 * F32);         // reused: conv1_raw
    float*    vv_out  = (float*)alloc((size_t)MROWS * 512 * F32);          // reused: conv2_raw
    _Float16* qn16    = (_Float16*)alloc((size_t)16 * 64 * NN_ * F16);     // reused: o16
    _Float16* kn16    = (_Float16*)alloc((size_t)16 * 64 * NN_ * F16);     // reused: skip32
    _Float16* k16     = (_Float16*)alloc((size_t)16 * 64 * NN_ * F16);     // reused: conv8_raw
    _Float16* qnT16   = (_Float16*)alloc((size_t)16 * NN_ * 64 * F16);     // reused: skip16
    _Float16* vca16   = (_Float16*)alloc((size_t)8 * 64 * NN_ * F16);
    _Float16* vsa16   = (_Float16*)alloc((size_t)8 * 64 * NN_ * F16);
    float*    kproj   = (float*)alloc((size_t)16 * 64 * 64 * F32);
    _Float16* kproj16 = (_Float16*)alloc((size_t)16 * 64 * 64 * F16);
    float*    vproj   = (float*)alloc((size_t)8 * 64 * 64 * F32);
    _Float16* vproj16 = (_Float16*)alloc((size_t)8 * 64 * 64 * F16);
    float*    aca     = (float*)alloc((size_t)16 * 64 * 64 * F32);
    _Float16* aca16   = (_Float16*)alloc((size_t)8 * 64 * 64 * F16);
    float*    asa     = (float*)alloc((size_t)16 * NN_ * 64 * F32);
    _Float16* asa16   = (_Float16*)alloc((size_t)8 * NN_ * 64 * F16);
    float*    xca     = (float*)alloc((size_t)8 * 64 * NN_ * F32);
    float*    xsa     = (float*)alloc((size_t)8 * NN_ * 64 * F32);
    _Float16* xca16   = (_Float16*)alloc((size_t)MROWS * 256 * F16);
    _Float16* xsa16   = (_Float16*)alloc((size_t)MROWS * 256 * F16);
    float*    outsa   = (float*)alloc((size_t)MROWS * 128 * F32);
    float*    outca   = (float*)alloc((size_t)MROWS * 128 * F32);
    // aliased late-stage buffers (lifetimes verified non-overlapping):
    _Float16* act1_16   = xln16;                 // [B,N,C] f16 channels-last
    float*    conv1_raw = qk_out;                // [B*N,256] f32
    float*    conv2_raw = vv_out;                // [B*N,256] f32
    _Float16* o16       = qn16;                  // [B,N,C] f16
    float*    skip32    = (float*)kn16;          // [B,C,N] f32 (8MB region)
    float*    conv8_raw = (float*)k16;           // [B*N,256] f32 (8MB region)
    _Float16* skip16    = qnT16;                 // [B,N,C] f16 channels-last (8MB region)

    auto cvt = [&](const float* s, _Float16* d, long long n) {
        cvt_f32_f16<<<dim3((unsigned)((n + 255) / 256)), 256, 0, stream>>>(s, d, n);
    };
    auto gemm_nt = [&](const _Float16* A, const _Float16* Bm, float* O, const float* bias,
                       int M, int Nc, int K, int batch,
                       long long sA, long long sB, long long sO) {
        const int tiles = (M >> 4) * (Nc >> 6);
        gemm_wmma<true><<<dim3((tiles + 3) / 4, batch), 128, 0, stream>>>(
            A, Bm, O, bias, M, Nc, K, sA, sB, sO);
    };
    auto gemm_nn = [&](const _Float16* A, const _Float16* Bm, float* O, const float* bias,
                       int M, int Nc, int K, int batch,
                       long long sA, long long sB, long long sO) {
        const int tiles = (M >> 4) * (Nc >> 6);
        gemm_wmma<false><<<dim3((tiles + 3) / 4, batch), 128, 0, stream>>>(
            A, Bm, O, bias, M, Nc, K, sA, sB, sO);
    };

    // ---- stage 0: weight conversion + lambda ----
    cvt(qk_w, w_qk16, 1024 * 256);
    cvt(vv_w, w_vv16, 512 * 256);
    cvt(E_w,  w_E16,  64 * 4096);
    cvt(op1_w, w_op1, 128 * 256);
    cvt(op2_w, w_op2, 128 * 256);
    {
        const long long ncw = (long long)CC * KCONV;
        cvt_convw<<<dim3((unsigned)((ncw + 255) / 256)), 256, 0, stream>>>(c1_w, w_c1);
        cvt_convw<<<dim3((unsigned)((ncw + 255) / 256)), 256, 0, stream>>>(c2_w, w_c2);
    }
    cvt(c8_w, w_c8, 256 * 256);
    lambda_kernel<<<1, 256, 0, stream>>>(lq1, lk1, lq2, lk2, lam);

    // ---- stage 1: layernorm, QKV projections ----
    ln_kernel<<<MROWS, 256, 0, stream>>>(x, ln_g, ln_b, xln16);
    gemm_nt(xln16, w_qk16, qk_out, nullptr, MROWS, 1024, 256, 1, 0, 0, 0);
    gemm_nt(xln16, w_vv16, vv_out, nullptr, MROWS, 512, 256, 1, 0, 0, 0);

    // ---- stage 2: l2norm / reshapes ----
    normqk_kernel<<<1024, 256, 0, stream>>>(qk_out, qn16, qnT16, kn16, k16);
    splitv_kernel<<<8192, 256, 0, stream>>>(vv_out, vca16, vsa16);

    // ---- stage 3: E projections (K = N = 4096) ----
    gemm_nt(k16,  w_E16, kproj, E_b, 64, 64, 4096, 16, (long long)64 * 4096, 0, 4096);
    cvt(kproj, kproj16, 16 * 64 * 64);
    gemm_nt(vsa16, w_E16, vproj, E_b, 64, 64, 4096, 8, (long long)64 * 4096, 0, 4096);
    cvt(vproj, vproj16, 8 * 64 * 64);

    // ---- stage 4: channel attention ----
    gemm_nt(qn16, kn16, aca, nullptr, 64, 64, 4096, 16,
            (long long)64 * 4096, (long long)64 * 4096, 4096);
    softmax_ca<<<512, 64, 0, stream>>>(aca, temp1, lam, aca16);
    gemm_nn(aca16, vca16, xca, nullptr, 64, 4096, 64, 8,
            4096, (long long)64 * 4096, (long long)64 * 4096);
    xpose_ca<<<8192, 256, 0, stream>>>(xca, xca16);

    // ---- stage 5: spatial attention ----
    gemm_nn(qnT16, kproj16, asa, nullptr, 4096, 64, 64, 16,
            (long long)4096 * 64, 4096, (long long)4096 * 64);
    softmax_sa<<<32768, 64, 0, stream>>>(asa, temp2, lam, asa16);
    gemm_nt(asa16, vproj16, xsa, nullptr, 4096, 64, 64, 8,
            (long long)4096 * 64, 4096, (long long)4096 * 64);
    xpose_sa<<<8192, 256, 0, stream>>>(xsa, xsa16);

    // ---- stage 6: output projections + residual ----
    gemm_nt(xsa16, w_op1, outsa, op1_b, MROWS, 128, 256, 1, 0, 0, 0);
    gemm_nt(xca16, w_op2, outca, op2_b, MROWS, 128, 256, 1, 0, 0, 0);
    combine_kernel<<<8192, 256, 0, stream>>>(x, outsa, outca, gamma, skip32, skip16);

    // ---- stage 7: UnetResBlock (conv-bn-lrelu, conv-bn, +skip, lrelu) ----
    conv3d_wmma<<<dim3((512 * 4) / 4), 128, 0, stream>>>(skip16, w_c1, c1_b, conv1_raw);
    bnstats_kernel<<<256, 256, 0, stream>>>(conv1_raw, bn1_m, bn1_r);
    bnact1_kernel<<<8192, 256, 0, stream>>>(conv1_raw, bn1_m, bn1_r, bn1_g, bn1_b, act1_16);
    conv3d_wmma<<<dim3((512 * 4) / 4), 128, 0, stream>>>(act1_16, w_c2, c2_b, conv2_raw);
    bnstats_kernel<<<256, 256, 0, stream>>>(conv2_raw, bn2_m, bn2_r);
    bnact2_kernel<<<8192, 256, 0, stream>>>(conv2_raw, bn2_m, bn2_r, bn2_g, bn2_b,
                                            skip32, o16);

    // ---- stage 8: 1x1x1 conv (GEMM) + final residual ----
    gemm_nt(o16, w_c8, conv8_raw, c8_b, MROWS, 256, 256, 1, 0, 0, 0);
    final_kernel<<<8192, 256, 0, stream>>>(skip32, conv8_raw, out);
}
